// MultiHeadAttention_13357348290609
// MI455X (gfx1250) — matrix operand order repaired and checked by reading, not compiled, not run on hardware
//
#include <hip/hip_runtime.h>
#include <hip/hip_bf16.h>

typedef __attribute__((ext_vector_type(16))) _Float16 v16h;
typedef __attribute__((ext_vector_type(8)))  _Float16 v8h;
typedef __attribute__((ext_vector_type(8)))  float    v8f;
typedef __attribute__((ext_vector_type(4)))  float    v4f;
typedef __attribute__((ext_vector_type(4)))  int      v4i;

static constexpr int BATCH = 2;
static constexpr int SLEN  = 4096;
static constexpr int DM    = 256;
static constexpr int KIN   = 264;          // k input feature dim (D_MODEL + K_MODE)
static constexpr int MROWS = BATCH * SLEN; // 8192

#if defined(__has_builtin)
#if __has_builtin(__builtin_amdgcn_global_load_async_to_lds_b128)
#define ASYNC_LDS 1
typedef __attribute__((address_space(1))) v4i gv4i;  // global int4
typedef __attribute__((address_space(3))) v4i sv4i;  // LDS int4
#endif
#endif

// ---------------------------------------------------------------------------
// 8-half vector load (with f32 -> f16 conversion overload)
// ---------------------------------------------------------------------------
__device__ __forceinline__ v8h ld8(const _Float16* p) { return *(const v8h*)p; }
__device__ __forceinline__ v8h ld8(const float* p) {
  v4f a = *(const v4f*)p, b = *(const v4f*)(p + 4);
  v8h r;
#pragma unroll
  for (int i = 0; i < 4; ++i) { r[i] = (_Float16)a[i]; r[4 + i] = (_Float16)b[i]; }
  return r;
}

// ---------------------------------------------------------------------------
// WMMA fragment loaders
// A-frag (16x32 f16): lane<16 -> row=lane, K=k0..k0+15
//                     lane>=16 -> row=lane-16, K=k0+16..k0+31
// B-frag (32x16):     lane<16 -> col=lane, K=k0..k0+15 ; lane>=16 -> col, K=k0+16..k0+31
// ---------------------------------------------------------------------------
template <typename T>
__device__ __forceinline__ v16h load_a_u(const T* __restrict__ rowp, int k0, int lane) {
  int koff = k0 + ((lane >> 4) << 4);
  v8h lo = ld8(rowp + koff), hi = ld8(rowp + koff + 8);
  v16h a;
#pragma unroll
  for (int i = 0; i < 8; ++i) { a[i] = lo[i]; a[8 + i] = hi[i]; }
  return a;
}

template <typename T>
__device__ __forceinline__ v16h load_b_u(const T* __restrict__ rowp, int k0, int lane) {
  int kb = k0 + ((lane >> 4) << 4);
  v8h lo = ld8(rowp + kb), hi = ld8(rowp + kb + 8);
  v16h b;
#pragma unroll
  for (int i = 0; i < 8; ++i) { b[i] = lo[i]; b[8 + i] = hi[i]; }
  return b;
}

template <typename T>
__device__ __forceinline__ v16h load_a_c(const T* __restrict__ rowp, int k0, int K, int lane) {
  int koff = k0 + ((lane >> 4) << 4);
  v16h a;
#pragma unroll
  for (int i = 0; i < 8; ++i) {
    int ka = koff + i, kb = koff + 8 + i;
    a[i]     = (ka < K) ? (_Float16)rowp[ka] : (_Float16)0.f;
    a[8 + i] = (kb < K) ? (_Float16)rowp[kb] : (_Float16)0.f;
  }
  return a;
}

template <typename T>
__device__ __forceinline__ v16h load_b_c(const T* __restrict__ rowp, int k0, int K, int lane) {
  int kb = k0 + ((lane >> 4) << 4);
  v16h b;
#pragma unroll
  for (int i = 0; i < 16; ++i) {
    int kk = kb + i;
    b[i] = (kk < K) ? (_Float16)rowp[kk] : (_Float16)0.f;
  }
  return b;
}

// ---------------------------------------------------------------------------
// Generic wave-tile GEMM:  C[M,N] = A[M,K] * W[N,K]^T + bias, K compile-time.
// One 16x16 output tile per wave, v_wmma_f32_16x16x32_f16 over K.
// ---------------------------------------------------------------------------
template <typename TA, typename TB, typename TC, int K>
__global__ void __launch_bounds__(256) gemm_wmma(const TA* __restrict__ A, int lda,
                                                 const TB* __restrict__ W, int ldw,
                                                 const float* __restrict__ bias,
                                                 TC* __restrict__ C, int ldc,
                                                 int M, int N) {
  const int lane   = threadIdx.x & 31;
  const int wid    = (blockIdx.x * blockDim.x + threadIdx.x) >> 5;
  const int tilesN = N >> 4;
  const int tm = wid / tilesN, tn = wid % tilesN;
  const int r0 = tm << 4;
  if (r0 >= M) return;
  const int n0   = tn << 4;
  const int lrow = lane & 15;

  const TA* arow = A + (size_t)(r0 + lrow) * lda;
  const TB* wrow = W + (size_t)(n0 + lrow) * ldw;

  v8f acc = {};
  constexpr int KF = K & ~31;
#pragma unroll
  for (int k0 = 0; k0 < KF; k0 += 32) {
    v16h a = load_a_u(arow, k0, lane);
    v16h b = load_b_u(wrow, k0, lane);
    acc = __builtin_amdgcn_wmma_f32_16x16x32_f16(false, a, false, b, (short)0, acc, false, false);
  }
  if constexpr ((K & 31) != 0) {  // checked tail (K=264 case)
    v16h a = load_a_c(arow, KF, K, lane);
    v16h b = load_b_c(wrow, KF, K, lane);
    acc = __builtin_amdgcn_wmma_f32_16x16x32_f16(false, a, false, b, (short)0, acc, false, false);
  }
  const float bv = bias ? bias[n0 + lrow] : 0.f;
  const int rowbase = r0 + ((lane >> 4) << 3);
#pragma unroll
  for (int r = 0; r < 8; ++r) {
    C[(size_t)(rowbase + r) * ldc + (n0 + lrow)] = (TC)(acc[r] + bv);
  }
}

// ---------------------------------------------------------------------------
// Wo_sum[n][d] = sum_h Wo[n][h*256 + d]  (exploits identical heads + tile)
// ---------------------------------------------------------------------------
__global__ void __launch_bounds__(256) wosum_kernel(const float* __restrict__ Wo,
                                                    _Float16* __restrict__ Wos) {
  int i = blockIdx.x * blockDim.x + threadIdx.x; // 0..65535
  int n = i >> 8, d = i & 255;
  float s = 0.f;
#pragma unroll
  for (int h = 0; h < 4; ++h) s += Wo[(size_t)n * 1024 + h * 256 + d];
  Wos[i] = (_Float16)s;
}

// ---------------------------------------------------------------------------
// Flash attention: per wave = 16 query rows, full 256-wide output in registers.
// Block = 8 waves = 128 query rows; K/V chunks (32 keys) staged in LDS per block
// (K chunk via async-to-LDS when available).
// ---------------------------------------------------------------------------
__global__ void __launch_bounds__(256, 1) attn_kernel(const _Float16* __restrict__ qh,
                                                      const _Float16* __restrict__ kh,
                                                      const _Float16* __restrict__ vh,
                                                      const int* __restrict__ mask,
                                                      _Float16* __restrict__ ph) {
  __shared__ __align__(16) _Float16 kh_s[32 * 264];   // keys x feat, stride 264 (conflict pad)
  __shared__ __align__(16) _Float16 vhT_s[256 * 40];  // feat x keys, stride 40 (conflict pad)
  __shared__ __align__(16) _Float16 p_s[8 * 512];     // per-wave 16x32 P transpose bounce

  const int tid   = threadIdx.x;
  const int lane  = tid & 31;
  const int wave  = tid >> 5;
  const int qrow0 = blockIdx.x * 128 + wave * 16;  // global row index (b*S + s)
  const int b     = qrow0 >> 12;
  const int lrow  = lane & 15;
  const int hb    = lane >> 4;

  // Preload Q fragments (16 rows x 256 feat as 8 A-frags)
  const _Float16* qp = qh + (size_t)(qrow0 + lrow) * DM;
  v16h qf[8];
#pragma unroll
  for (int j = 0; j < 8; ++j) {
    int koff = 32 * j + (hb << 4);
    v8h lo = *(const v8h*)(qp + koff);
    v8h hi = *(const v8h*)(qp + koff + 8);
#pragma unroll
    for (int i = 0; i < 8; ++i) { qf[j][i] = lo[i]; qf[j][8 + i] = hi[i]; }
  }

  v8f acc[16] = {};
  float mrow[8], lsum[8];
#pragma unroll
  for (int r = 0; r < 8; ++r) { mrow[r] = -3.0e38f; lsum[r] = 0.f; }

  const float scale   = 0.0625f;  // 1/sqrt(256)
  const int   stage_r = tid >> 3;
  const int   stage_c = (tid & 7) * 32;
  _Float16*   pw      = p_s + wave * 512;
  const int*  mwave   = mask + (size_t)(qrow0 + hb * 8) * SLEN + lrow;

  for (int k0 = 0; k0 < SLEN; k0 += 32) {
    __syncthreads();
    {  // stage K chunk (row-major, async if available) and V chunk (transposed)
      const _Float16* ks = kh + (size_t)(b * SLEN + k0 + stage_r) * DM + stage_c;
      const _Float16* vs = vh + (size_t)(b * SLEN + k0 + stage_r) * DM + stage_c;
      _Float16*       kd = kh_s + stage_r * 264 + stage_c;
#pragma unroll
      for (int u = 0; u < 4; ++u) {
#ifdef ASYNC_LDS
        __builtin_amdgcn_global_load_async_to_lds_b128(
            (gv4i*)(ks + u * 8), (sv4i*)(kd + u * 8), 0, 0);
#else
        *(v8h*)(kd + u * 8) = *(const v8h*)(ks + u * 8);
#endif
        v8h vv = *(const v8h*)(vs + u * 8);
#pragma unroll
        for (int i = 0; i < 8; ++i) vhT_s[(stage_c + u * 8 + i) * 40 + stage_r] = vv[i];
      }
#ifdef ASYNC_LDS
#if __has_builtin(__builtin_amdgcn_s_wait_asynccnt)
      __builtin_amdgcn_s_wait_asynccnt(0);
#else
      asm volatile("s_wait_asynccnt 0" ::: "memory");
#endif
#endif
    }
    __syncthreads();

    // ---- hoist mask loads so HBM latency hides under the score WMMAs ----
    int mk[16];
    {
      const int* mp = mwave + k0;
#pragma unroll
      for (int t2 = 0; t2 < 2; ++t2)
#pragma unroll
        for (int r = 0; r < 8; ++r) mk[t2 * 8 + r] = mp[(size_t)r * SLEN + t2 * 16];
    }

    // ---- scores for 32 keys: two 16x16 tiles via 8 WMMAs each ----
    float sv[16];
#pragma unroll
    for (int t2 = 0; t2 < 2; ++t2) {
      v8f sc = {};
      const _Float16* krow = kh_s + (t2 * 16 + lrow) * 264;
#pragma unroll
      for (int j = 0; j < 8; ++j) {
        int kb = 32 * j + (hb << 4);
        v8h b0 = *(const v8h*)(krow + kb);
        v8h b1 = *(const v8h*)(krow + kb + 8);
        v16h bf;
#pragma unroll
        for (int i = 0; i < 8; ++i) { bf[i] = b0[i]; bf[8 + i] = b1[i]; }
        sc = __builtin_amdgcn_wmma_f32_16x16x32_f16(false, qf[j], false, bf, (short)0, sc, false, false);
      }
#pragma unroll
      for (int r = 0; r < 8; ++r)
        sv[t2 * 8 + r] = (mk[t2 * 8 + r] == 0) ? -1.0e9f : sc[r] * scale;
    }

    // ---- online softmax (row reductions across 16-lane halves) ----
#pragma unroll
    for (int r = 0; r < 8; ++r) {
      float cm = fmaxf(sv[r], sv[8 + r]);
#pragma unroll
      for (int off = 8; off >= 1; off >>= 1) cm = fmaxf(cm, __shfl_xor(cm, off, 32));
      float mnew  = fmaxf(mrow[r], cm);
      float alpha = __expf(mrow[r] - mnew);
      mrow[r] = mnew;
      float p0 = __expf(sv[r] - mnew);
      float p1 = __expf(sv[8 + r] - mnew);
      sv[r] = p0; sv[8 + r] = p1;
      float ps = p0 + p1;
#pragma unroll
      for (int off = 8; off >= 1; off >>= 1) ps += __shfl_xor(ps, off, 32);
      lsum[r] = lsum[r] * alpha + ps;
#pragma unroll
      for (int n = 0; n < 16; ++n) acc[n][r] *= alpha;
    }

    // ---- P: C-layout -> A-layout via per-wave LDS bounce ----
#pragma unroll
    for (int r = 0; r < 8; ++r) {
      pw[(hb * 8 + r) * 32 + lrow]      = (_Float16)sv[r];
      pw[(hb * 8 + r) * 32 + 16 + lrow] = (_Float16)sv[8 + r];
    }
    asm volatile("s_wait_dscnt 0" ::: "memory");  // same-wave DS RAW fence
    v16h pa;
    {
      v8h a0 = *(const v8h*)(pw + lrow * 32 + (hb << 4));
      v8h a1 = *(const v8h*)(pw + lrow * 32 + 8 + (hb << 4));
#pragma unroll
      for (int i = 0; i < 8; ++i) { pa[i] = a0[i]; pa[8 + i] = a1[i]; }
    }

    // ---- PV: 16 output tiles, one WMMA each (k=32) ----
#pragma unroll
    for (int n = 0; n < 16; ++n) {
      const _Float16* vrow = vhT_s + (n * 16 + lrow) * 40 + (hb << 4);
      v8h b0 = *(const v8h*)(vrow);
      v8h b1 = *(const v8h*)(vrow + 8);
      v16h bf;
#pragma unroll
      for (int i = 0; i < 8; ++i) { bf[i] = b0[i]; bf[8 + i] = b1[i]; }
      acc[n] = __builtin_amdgcn_wmma_f32_16x16x32_f16(false, pa, false, bf, (short)0, acc[n], false, false);
    }

    if (k0 + 32 < SLEN)
      __builtin_prefetch(mwave + k0 + 32, 0, 1);
  }

  // ---- finalize: divide by row sums, store f16 per_head ----
#pragma unroll
  for (int r = 0; r < 8; ++r) {
    float inv = 1.f / lsum[r];
    _Float16* op = ph + (size_t)(qrow0 + hb * 8 + r) * DM + lrow;
#pragma unroll
    for (int n = 0; n < 16; ++n) op[n * 16] = (_Float16)(acc[n][r] * inv);
  }
}

// ---------------------------------------------------------------------------
extern "C" void kernel_launch(void* const* d_in, const int* in_sizes, int n_in,
                              void* d_out, int out_size, void* d_ws, size_t ws_size,
                              hipStream_t stream) {
  const float* q    = (const float*)d_in[0];
  const float* k    = (const float*)d_in[1];
  const float* v    = (const float*)d_in[2];
  const int*   mask = (const int*)d_in[3];
  const float* Wq   = (const float*)d_in[4];
  const float* bq   = (const float*)d_in[5];
  const float* Wk   = (const float*)d_in[6];
  const float* bk   = (const float*)d_in[7];
  const float* Wv   = (const float*)d_in[8];
  const float* bv   = (const float*)d_in[9];
  const float* Wo   = (const float*)d_in[10];
  const float* bo   = (const float*)d_in[11];
  float*       out  = (float*)d_out;

  char* ws = (char*)d_ws;
  _Float16* qh  = (_Float16*)(ws);                        // 4 MB
  _Float16* kh  = (_Float16*)(ws + ((size_t)4  << 20));   // 4 MB
  _Float16* vh  = (_Float16*)(ws + ((size_t)8  << 20));   // 4 MB
  _Float16* phd = (_Float16*)(ws + ((size_t)12 << 20));   // 4 MB
  _Float16* wos = (_Float16*)(ws + ((size_t)16 << 20));   // 128 KB

  dim3 blk(256);
  // (M/16)*(N/16) = 8192 wave-tiles / 8 waves per block = 1024 blocks
  gemm_wmma<float, float, _Float16, DM> <<<1024, blk, 0, stream>>>(q, DM,  Wq, DM,  bq, qh, DM, MROWS, DM);
  gemm_wmma<float, float, _Float16, KIN><<<1024, blk, 0, stream>>>(k, KIN, Wk, KIN, bk, kh, DM, MROWS, DM);
  gemm_wmma<float, float, _Float16, DM> <<<1024, blk, 0, stream>>>(v, DM,  Wv, DM,  bv, vh, DM, MROWS, DM);
  wosum_kernel<<<256, blk, 0, stream>>>(Wo, wos);
  attn_kernel<<<64, blk, 0, stream>>>(qh, kh, vh, mask, phd);
  gemm_wmma<_Float16, _Float16, float, DM><<<1024, blk, 0, stream>>>(phd, DM, wos, DM, bo, out, DM, MROWS, DM);
}
